// RNNBase_54331336294826
// MI455X (gfx1250) — matrix operand order repaired and checked by reading, not compiled, not run on hardware
//
#include <hip/hip_runtime.h>
#include <hip/hip_bf16.h>
#include <math.h>

// ---------------------------------------------------------------------------
// 2-layer vanilla RNN LM forward on MI455X (gfx1250, wave32, WMMA + TDM).
//   S=64 B=64 V=10000 E=256 H=512 L=2
//   1) weights/hidden -> f16; gather embeddings -> f16
//   2) A0 = X0 @ Wx0^T + bh0           (4096x512x256 WMMA GEMM)
//   3) 64 sequential steps x 3 launches (64x512x512 WMMA GEMMs)
//   4) logits = H1_all @ Wy1^T + by1   (4096x10000x512 WMMA GEMM, dominant)
//      B slab staged to LDS via Tensor Data Mover (pad-inserted, bank-friendly)
// ---------------------------------------------------------------------------

typedef __attribute__((ext_vector_type(16))) _Float16 v16h;
typedef __attribute__((ext_vector_type(8)))  _Float16 v8h;
typedef __attribute__((ext_vector_type(8)))  float    v8f;
typedef __attribute__((ext_vector_type(4)))  unsigned int v4u;
typedef __attribute__((ext_vector_type(8)))  int      v8i;
typedef __attribute__((ext_vector_type(4)))  int      v4i;

#define SEQ    64
#define BATCH  64
#define VOCAB  10000
#define EDIM   256
#define HDIM   512
#define SB     (SEQ * BATCH)   // 4096

#if defined(__has_builtin)
#  if __has_builtin(__builtin_amdgcn_tensor_load_to_lds) && \
      __has_builtin(__builtin_amdgcn_s_wait_tensorcnt)
#    define USE_TDM 1
#  endif
#endif

// ---- WMMA fragment loaders (contiguous k order, same in A and B, wave32) ----------
// A (16x32 f16, MxK): lane l<16 -> row l, K {kb+0..15};
//                     lane l>=16 -> row l-16, K {kb+16..31}.
static __device__ __forceinline__
v16h load_a_frag(const _Float16* __restrict__ A, int lda, int mBase, int kBase, int lane)
{
    int row = mBase + (lane & 15);
    int hi8 = (lane >> 4) << 4;
    const _Float16* p = A + (size_t)row * lda + kBase + hi8;
    v8h a0 = *(const v8h*)(p);
    v8h a1 = *(const v8h*)(p + 8);
    return __builtin_shufflevector(a0, a1, 0,1,2,3,4,5,6,7,8,9,10,11,12,13,14,15);
}

// B (32x16 f16, KxN) with B[k][n] = W[n][k], W row-major (N,K):
// lane l<16: col nb+l, K {kb..kb+15}; lane l>=16: col nb+l-16, K {kb+16..kb+31}.
static __device__ __forceinline__
v16h load_b_frag(const _Float16* __restrict__ W, int ldk, int nBase, int kBase, int lane)
{
    int col  = nBase + (lane & 15);
    int hi16 = (lane >> 4) << 4;
    const _Float16* p = W + (size_t)col * ldk + kBase + hi16;
    v8h b0 = *(const v8h*)(p);
    v8h b1 = *(const v8h*)(p + 8);
    return __builtin_shufflevector(b0, b1, 0,1,2,3,4,5,6,7,8,9,10,11,12,13,14,15);
}

// ---- generic WMMA GEMM: out = act( A(MxK,f16) @ W(NxK,f16)^T + Cadd + bias )
__global__ __launch_bounds__(256)
void gemm16(const _Float16* __restrict__ A, const _Float16* __restrict__ W,
            const float* __restrict__ Cadd, const float* __restrict__ bias,
            float* __restrict__ outF, _Float16* __restrict__ outH,
            int M, int N, int K, int doTanh)
{
    int lane = threadIdx.x & 31;
    int wave = threadIdx.x >> 5;
    int tile = blockIdx.x * (blockDim.x >> 5) + wave;
    int nT   = N >> 4;
    if (tile >= (M >> 4) * nT) return;                // wave-uniform exit
    int mBase = (tile / nT) << 4;
    int nBase = (tile % nT) << 4;

    v8f acc = {0.f,0.f,0.f,0.f,0.f,0.f,0.f,0.f};
    for (int k = 0; k < K; k += 32) {
        v16h a = load_a_frag(A, K, mBase, k, lane);
        v16h b = load_b_frag(W, K, nBase, k, lane);
        acc = __builtin_amdgcn_wmma_f32_16x16x32_f16(false, a, false, b,
                                                     (short)0, acc, false, false);
    }

    int col  = nBase + (lane & 15);
    int rOff = mBase + ((lane >> 4) << 3);
    float bv = bias ? bias[col] : 0.0f;
#pragma unroll
    for (int r = 0; r < 8; ++r) {
        size_t idx = (size_t)(rOff + r) * N + col;
        float v = acc[r] + bv;
        if (Cadd)   v += Cadd[idx];
        if (doTanh) v  = tanhf(v);
        if (outF) outF[idx] = v;
        if (outH) outH[idx] = (_Float16)v;
    }
}

// ---- fused mid-step kernel: tiles 0..127 -> x1 = h0@Wy0^T + by0 (f16 out)
//                             tiles 128..255 -> p  = hw@Wh1^T        (f32 out)
__global__ __launch_bounds__(256)
void step2_fused(const _Float16* __restrict__ h0, const _Float16* __restrict__ hw,
                 const _Float16* __restrict__ Wy0h, const _Float16* __restrict__ Wh1h,
                 const float* __restrict__ by0,
                 _Float16* __restrict__ x1, float* __restrict__ p)
{
    int lane  = threadIdx.x & 31;
    int wave  = threadIdx.x >> 5;
    int tile  = blockIdx.x * 8 + wave;               // 256 tiles
    int which = tile >> 7;                           // wave-uniform
    int t2    = tile & 127;
    int mBase = (t2 >> 5) << 4;                      // 4 M tiles
    int nBase = (t2 & 31) << 4;                      // 32 N tiles
    const _Float16* A = which ? hw  : h0;
    const _Float16* W = which ? Wh1h : Wy0h;

    v8f acc = {0.f,0.f,0.f,0.f,0.f,0.f,0.f,0.f};
    for (int k = 0; k < HDIM; k += 32) {
        v16h a = load_a_frag(A, HDIM, mBase, k, lane);
        v16h b = load_b_frag(W, HDIM, nBase, k, lane);
        acc = __builtin_amdgcn_wmma_f32_16x16x32_f16(false, a, false, b,
                                                     (short)0, acc, false, false);
    }
    int col  = nBase + (lane & 15);
    int rOff = mBase + ((lane >> 4) << 3);
    if (which == 0) {
        float bv = by0[col];
#pragma unroll
        for (int r = 0; r < 8; ++r)
            x1[(size_t)(rOff + r) * HDIM + col] = (_Float16)(acc[r] + bv);
    } else {
#pragma unroll
        for (int r = 0; r < 8; ++r)
            p[(size_t)(rOff + r) * HDIM + col] = acc[r];
    }
}

// ---- logits GEMM: 4096 x 10000 x 512 ---------------------------------------
// Block = 8 M-waves x 80 N-cols. B slab staged to LDS in K-chunks of 128
// (rows padded by 8 halves -> 272B stride -> conflict-free ds_load_b128).
// Staging uses the Tensor Data Mover with hardware pad insertion when present.
#define NT      5                 // 5 N tiles (80 cols) per wave
#define KCH     128               // K chunk staged per iteration
#define LDST    (KCH + 8)         // padded LDS row stride in halves (272B)

__global__ __launch_bounds__(256)
void gemm_logits(const _Float16* __restrict__ A, const _Float16* __restrict__ W,
                 const float* __restrict__ bias, float* __restrict__ out)
{
    __shared__ __align__(16) _Float16 Bsh[80 * LDST];   // ~21.3 KB

    int lane  = threadIdx.x & 31;
    int wave  = threadIdx.x >> 5;
    int mBase = (blockIdx.x * 8 + wave) << 4;
    int nBase = blockIdx.y * (16 * NT);

    v8f acc[NT];
#pragma unroll
    for (int t = 0; t < NT; ++t) acc[t] = (v8f){0.f,0.f,0.f,0.f,0.f,0.f,0.f,0.f};

    for (int kc = 0; kc < HDIM; kc += KCH) {
        __syncthreads();                              // previous chunk consumed
#if defined(USE_TDM)
        if (wave == 0) {
            // D# for a 2D tile: 32 x 80 elements of 8B (=256B rows, 1024B stride),
            // pad 4 DWORDs after every 64 DWORDs -> LDS row stride 272B.
            unsigned long long ga = (unsigned long long)(const void*)W
                                  + (unsigned long long)nBase * (HDIM * 2)
                                  + (unsigned long long)kc * 2;
            // LDS byte offset = low 32 bits of the LDS-aperture flat address
            unsigned int lds = (unsigned int)(unsigned long long)(const void*)Bsh;
            v4u g0; v8i g1; v4i gz;
            g0[0] = 1u;                               // count=1
            g0[1] = lds;                              // lds_addr
            g0[2] = (unsigned int)(ga & 0xffffffffu);
            g0[3] = (unsigned int)((ga >> 32) & 0x1ffffffu) | (2u << 30); // type=2
            g1[0] = (int)((3u << 16) | (1u << 20) | (5u << 22) | (3u << 25));
            g1[1] = (int)(32u << 16);                 // tensor_dim0 = 32
            g1[2] = (int)(80u << 16);                 // tensor_dim1 = 80
            g1[3] = (int)(32u << 16);                 // tile_dim0   = 32
            g1[4] = (int)80u;                         // tile_dim1   = 80
            g1[5] = (int)128u;                        // tensor_dim0_stride = 128 elems
            g1[6] = 0; g1[7] = 0;
            gz[0] = gz[1] = gz[2] = gz[3] = 0;
#  if __clang_major__ >= 23
            v8i gz8; for (int z = 0; z < 8; ++z) gz8[z] = 0;
            __builtin_amdgcn_tensor_load_to_lds(g0, g1, gz, gz, gz8, 0);
#  else
            __builtin_amdgcn_tensor_load_to_lds(g0, g1, gz, gz, 0);
#  endif
            __builtin_amdgcn_s_wait_tensorcnt(0);
        }
#else
        // cooperative staging fallback: 80 rows x 128 halves, padded stride
        for (int i = threadIdx.x; i < 80 * (KCH / 8); i += 256) {
            int r = i >> 4, c = i & 15;               // 16 x v8h per row
            *(v8h*)(&Bsh[r * LDST + c * 8]) =
                *(const v8h*)(W + (size_t)(nBase + r) * HDIM + kc + c * 8);
        }
#endif
        __syncthreads();

#pragma unroll
        for (int k4 = 0; k4 < KCH; k4 += 32) {
            v16h a = load_a_frag(A, HDIM, mBase, kc + k4, lane);
            int hi16 = (lane >> 4) << 4;
#pragma unroll
            for (int t = 0; t < NT; ++t) {
                int colLocal = 16 * t + (lane & 15);
                const _Float16* bp = &Bsh[colLocal * LDST + k4 + hi16];
                v8h b0 = *(const v8h*)(bp);
                v8h b1 = *(const v8h*)(bp + 8);
                v16h b = __builtin_shufflevector(b0, b1,
                         0,1,2,3,4,5,6,7,8,9,10,11,12,13,14,15);
                acc[t] = __builtin_amdgcn_wmma_f32_16x16x32_f16(false, a, false, b,
                                                                (short)0, acc[t],
                                                                false, false);
            }
        }
    }

    int rOff = mBase + ((lane >> 4) << 3);
#pragma unroll
    for (int t = 0; t < NT; ++t) {
        int col  = nBase + 16 * t + (lane & 15);
        float bv = bias[col];
#pragma unroll
        for (int r = 0; r < 8; ++r)
            out[(size_t)(rOff + r) * VOCAB + col] = acc[t][r] + bv;
    }
}

// ---- helpers ---------------------------------------------------------------
__global__ void f2h_kernel(const float* __restrict__ s, _Float16* __restrict__ d, int n)
{
    int i = blockIdx.x * blockDim.x + threadIdx.x;
    if (i < n) d[i] = (_Float16)s[i];
}

__global__ void embed_gather(const int* __restrict__ idx, const float* __restrict__ table,
                             _Float16* __restrict__ X)
{
    int i = blockIdx.x * blockDim.x + threadIdx.x;
    if (i >= SB * EDIM) return;
    int row = i / EDIM, e = i - row * EDIM;
    X[i] = (_Float16)table[(size_t)idx[row] * EDIM + e];
}

// ---------------------------------------------------------------------------
extern "C" void kernel_launch(void* const* d_in, const int* in_sizes, int n_in,
                              void* d_out, int out_size, void* d_ws, size_t ws_size,
                              hipStream_t stream)
{
    (void)in_sizes; (void)n_in; (void)out_size; (void)ws_size;

    const int*   inputs = (const int*)  d_in[0];
    const float* hidden = (const float*)d_in[1];
    const float* emb    = (const float*)d_in[2];
    const float* Wx0    = (const float*)d_in[3];
    const float* Wh0    = (const float*)d_in[4];
    const float* bh0    = (const float*)d_in[5];
    const float* Wy0    = (const float*)d_in[6];
    const float* by0    = (const float*)d_in[7];
    const float* Wx1    = (const float*)d_in[8];
    const float* Wh1    = (const float*)d_in[9];
    const float* bh1    = (const float*)d_in[10];
    const float* Wy1    = (const float*)d_in[11];
    const float* by1    = (const float*)d_in[12];

    float* logits = (float*)d_out;
    float* states = logits + (size_t)SEQ * BATCH * VOCAB;

    char* ws = (char*)d_ws;
    size_t off = 0;
    auto take = [&](size_t bytes) -> void* {
        void* p = ws + off;
        off = (off + bytes + 255) & ~(size_t)255;
        return p;
    };
    _Float16* wx0h  = (_Float16*)take((size_t)HDIM * EDIM * 2);
    _Float16* wh0h  = (_Float16*)take((size_t)HDIM * HDIM * 2);
    _Float16* wy0h  = (_Float16*)take((size_t)HDIM * HDIM * 2);
    _Float16* wx1h  = (_Float16*)take((size_t)HDIM * HDIM * 2);
    _Float16* wh1h  = (_Float16*)take((size_t)HDIM * HDIM * 2);
    _Float16* wy1h  = (_Float16*)take((size_t)VOCAB * HDIM * 2);
    _Float16* x0h   = (_Float16*)take((size_t)SB * EDIM * 2);
    _Float16* h1all = (_Float16*)take((size_t)SB * HDIM * 2);
    float*    a0    = (float*)   take((size_t)SB * HDIM * 4);
    _Float16* h0h   = (_Float16*)take((size_t)BATCH * HDIM * 2);
    _Float16* x1h   = (_Float16*)take((size_t)BATCH * HDIM * 2);
    float*    pacc  = (float*)   take((size_t)BATCH * HDIM * 4);
    _Float16* hid0h = (_Float16*)take((size_t)BATCH * HDIM * 2);
    _Float16* hid1h = (_Float16*)take((size_t)BATCH * HDIM * 2);
    float*    stScr = (float*)   take((size_t)BATCH * HDIM * 4);

    auto cvt = [&](const float* s, _Float16* d, int n) {
        f2h_kernel<<<(n + 255) / 256, 256, 0, stream>>>(s, d, n);
    };

    cvt(Wx0, wx0h, HDIM * EDIM);
    cvt(Wh0, wh0h, HDIM * HDIM);
    cvt(Wy0, wy0h, HDIM * HDIM);
    cvt(Wx1, wx1h, HDIM * HDIM);
    cvt(Wh1, wh1h, HDIM * HDIM);
    cvt(Wy1, wy1h, VOCAB * HDIM);
    cvt(hidden,                hid0h, BATCH * HDIM);
    cvt(hidden + BATCH * HDIM, hid1h, BATCH * HDIM);
    embed_gather<<<(SB * EDIM + 255) / 256, 256, 0, stream>>>(inputs, emb, x0h);

    // A0 = X0 @ Wx0^T + bh0
    {
        int tiles = (SB / 16) * (HDIM / 16);
        gemm16<<<(tiles + 7) / 8, 256, 0, stream>>>(
            x0h, wx0h, nullptr, bh0, a0, nullptr, SB, HDIM, EDIM, 0);
    }

    // sequential recurrence
    const int recBlocks = ((BATCH / 16) * (HDIM / 16)) / 8;   // 16
    for (int t = 0; t < SEQ; ++t) {
        const _Float16* hin = (t == 0) ? hid0h
                                       : h1all + (size_t)(t - 1) * BATCH * HDIM;
        float* st0 = (t == SEQ - 1) ? states : stScr;
        float* st1 = (t == SEQ - 1) ? states + BATCH * HDIM : stScr;

        // h0 = tanh(A0[t] + hin @ Wh0^T)
        gemm16<<<recBlocks, 256, 0, stream>>>(
            hin, wh0h, a0 + (size_t)t * BATCH * HDIM, nullptr,
            st0, h0h, BATCH, HDIM, HDIM, 1);

        // fused: x1 = h0 @ Wy0^T + by0 ; pacc = hprev1 @ Wh1^T
        const _Float16* hw = (t == 0) ? hid1h : h0h;
        step2_fused<<<32, 256, 0, stream>>>(h0h, hw, wy0h, wh1h, by0, x1h, pacc);

        // h1 = tanh(x1 @ Wx1^T + pacc + bh1) -> H1_all[t]
        gemm16<<<recBlocks, 256, 0, stream>>>(
            x1h, wx1h, pacc, bh1, st1,
            h1all + (size_t)t * BATCH * HDIM, BATCH, HDIM, HDIM, 1);
    }

    // logits = H1_all @ Wy1^T + by1
    gemm_logits<<<dim3(SB / 128, VOCAB / 80), 256, 0, stream>>>(
        h1all, wy1h, by1, logits);
}
